// ResidualVQ_63866163692082
// MI455X (gfx1250) — matrix operand order repaired and checked by reading, not compiled, not run on hardware
//
#include <hip/hip_runtime.h>

#define Bb   32
#define Dd   512
#define Tt   2048
#define Vv   1024
#define Qq   6
#define Nn   (Bb * Tt)          // 65536 rows
#define BETA 0.25f
#define NBLK (Nn / 32)          // 2048 main-kernel blocks

typedef __attribute__((ext_vector_type(16))) __bf16 v16bf;
typedef __attribute__((ext_vector_type(8)))  __bf16 v8bf;
typedef __attribute__((ext_vector_type(4)))  __bf16 v4bf;
typedef __attribute__((ext_vector_type(8)))  float  v8f;
typedef __attribute__((ext_vector_type(4)))  float  v4f;

// -------------------------------------------------------------------------
// Prep: per-code squared norms ||c||^2 and (optionally) bf16 codebook copy.
// One block per (q, code) row, 128 threads, 4 dims/thread.
// -------------------------------------------------------------------------
__global__ __launch_bounds__(128)
void rvq_prep(const float* __restrict__ cb, float* __restrict__ c2,
              __bf16* __restrict__ cbBF, int writeBF) {
    __shared__ float sh[4];
    const int code = blockIdx.x;                 // 0 .. Q*V-1
    const int tid  = threadIdx.x;                // 128 threads
    const float* row = cb + (size_t)code * Dd;
    v4f f = ((const v4f*)row)[tid];              // dims tid*4 .. tid*4+3
    if (writeBF) {
        v4bf h;
        h[0] = (__bf16)f[0]; h[1] = (__bf16)f[1];
        h[2] = (__bf16)f[2]; h[3] = (__bf16)f[3];
        *(v4bf*)(cbBF + (size_t)code * Dd + tid * 4) = h;
    }
    float s = f[0]*f[0] + f[1]*f[1] + f[2]*f[2] + f[3]*f[3];
    #pragma unroll
    for (int off = 16; off >= 1; off >>= 1) s += __shfl_xor(s, off, 32);
    if ((tid & 31) == 0) sh[tid >> 5] = s;
    __syncthreads();
    if (tid == 0) c2[code] = sh[0] + sh[1] + sh[2] + sh[3];
}

__global__ __launch_bounds__(256)
void rvq_zero(float* __restrict__ p, int n) {
    int i = blockIdx.x * 256 + threadIdx.x;
    if (i < n) p[i] = 0.0f;
}

// -------------------------------------------------------------------------
// Main fused kernel: 256 threads (8 waves) per block, 32 rows per block.
// Residual kept exact in fp32 registers; bf16 shadow in LDS feeds WMMA A.
// bf16 path: A fragments for the whole K dimension are explicitly hoisted
// into registers once per quantizer (they are invariant across the 4 column
// groups), and B fragments stream from the bf16 codebook through a 3-stage
// register pipeline (issued 2 k-steps ahead -> staggered loadcnt waits).
// No min-occupancy cap: ~450 VGPRs, 2 waves/SIMD, zero scratch spills.
// -------------------------------------------------------------------------
template <bool USE_BF16CB>
__global__ __launch_bounds__(256)
void rvq_main(const float*  __restrict__ x,
              const float*  __restrict__ cb,
              const __bf16* __restrict__ cbBF,
              const float*  __restrict__ c2,
              float* __restrict__ outQ,      // (B, D, T) quantized_out
              float* __restrict__ outIdx,    // (B, T, Q) indices as float
              float* __restrict__ lossPart,  // [Q * NBLK] deterministic partials
              float* __restrict__ counts) {  // [Q * V] histogram (+1.0 exact)
    __shared__ __bf16 resBF[32][520];        // padded: conflict-light b128 reads
    __shared__ float  candV[8][32];
    __shared__ int    candI[8][32];
    __shared__ int    idxLDS[32];
    __shared__ float  lossW[8];

    const int tid  = threadIdx.x;
    const int lane = tid & 31;
    const int wv   = tid >> 5;               // wave 0..7
    const int r    = tid & 31;               // owned row (data phases)
    const int g    = tid >> 5;               // owned dim group (data phases)
    const int n0   = blockIdx.x * 32;        // first flat row (b*T + t)
    const int bIdx = n0 / Tt;
    const int t0   = n0 % Tt;                // rows of a block share b (T%32==0)

    // ---- load x -> fp32 residual registers + bf16 LDS copy ----
    float resR[64];
    {
        const float* xb = x + (size_t)bIdx * Dd * Tt + t0 + r;
        #pragma unroll
        for (int i = 0; i < 64; ++i) {
            const int d = g + (i << 3);
            const float v = xb[(size_t)d * Tt];
            resR[i] = v;
            resBF[r][d] = (__bf16)v;
        }
    }
    __syncthreads();

    const int halfSel = lane >> 4;           // 0: lanes 0-15, 1: lanes 16-31
    const int koffA   = halfSel * 16;        // A: K {koffA..+15} contiguous
    const int koffB   = halfSel * 16;        // B: K {koffB..+15} contiguous
    const int laneN   = lane & 15;
    const int col0    = wv * 128;            // 128 codes per wave

    for (int q = 0; q < Qq; ++q) {
        const float*  cbq   = cb   + (size_t)q * Vv * Dd;
        const __bf16* cbBFq = cbBF + (size_t)q * Vv * Dd;
        const float*  c2q   = c2 + q * Vv;

        float minv[2][8];
        int   mini[2][8];
        #pragma unroll
        for (int rt = 0; rt < 2; ++rt)
            #pragma unroll
            for (int j = 0; j < 8; ++j) { minv[rt][j] = 3.4028235e38f; mini[rt][j] = 0; }

        if constexpr (USE_BF16CB) {
            // ---- explicit hoist: all A fragments for K=512 (invariant/ctg) ----
            v16bf afr[16][2];                        // 256 VGPRs, held for this q
            #pragma unroll
            for (int kc = 0; kc < 16; ++kc) {
                const int d0 = kc * 32;
                #pragma unroll
                for (int rt = 0; rt < 2; ++rt) {
                    const int rowA = rt * 16 + laneN;
                    v8bf alo = *(const v8bf*)&resBF[rowA][d0 + koffA];
                    v8bf ahi = *(const v8bf*)&resBF[rowA][d0 + koffA + 8];
                    afr[kc][rt] = __builtin_shufflevector(alo, ahi,
                                  0,1,2,3,4,5,6,7,8,9,10,11,12,13,14,15);
                }
            }
            // ---- B stream + WMMA: 4 column groups of 2 tiles ----
            for (int ctg = 0; ctg < 4; ++ctg) {
                v8f acc[2][2] = {};                  // [row-tile][col-tile]
                const int colBase = col0 + ctg * 32;
                const __bf16* bb0 = cbBFq + (size_t)(colBase + laneN) * Dd + koffB;
                const __bf16* bb1 = cbBFq + (size_t)(colBase + 16 + laneN) * Dd + koffB;
                v16bf bbuf[3][2];                    // 3-stage rotating B pipeline
                #pragma unroll
                for (int p = 0; p < 2; ++p) {        // prologue: stages 0,1
                    bbuf[p][0] = *(const v16bf*)(bb0 + p * 32);
                    bbuf[p][1] = *(const v16bf*)(bb1 + p * 32);
                }
                #pragma unroll
                for (int kc = 0; kc < 16; ++kc) {    // K = 512 in steps of 32
                    if (kc < 14) {                   // prefetch 2 k-steps ahead
                        bbuf[(kc + 2) % 3][0] = *(const v16bf*)(bb0 + (kc + 2) * 32);
                        bbuf[(kc + 2) % 3][1] = *(const v16bf*)(bb1 + (kc + 2) * 32);
                    }
                    #pragma unroll
                    for (int ct = 0; ct < 2; ++ct)
                        #pragma unroll
                        for (int rt = 0; rt < 2; ++rt)
                            acc[rt][ct] = __builtin_amdgcn_wmma_f32_16x16x32_bf16(
                                false, afr[kc][rt], false, bbuf[kc % 3][ct], (short)0,
                                acc[rt][ct], false, false);
                }
                // fold: dist = ||c||^2 - 2*dot  (||r||^2 constant per row)
                #pragma unroll
                for (int ct = 0; ct < 2; ++ct) {
                    const int cA = colBase + ct * 16 + laneN;
                    const float c2v = c2q[cA];
                    #pragma unroll
                    for (int rt = 0; rt < 2; ++rt)
                        #pragma unroll
                        for (int j = 0; j < 8; ++j) {
                            const float dval = c2v - 2.0f * acc[rt][ct][j];
                            if (dval < minv[rt][j]) { minv[rt][j] = dval; mini[rt][j] = cA; }
                        }
                }
            }
        } else {
            // ---- fallback: fp32 codebook loads, convert in the loop ----
            for (int ctg = 0; ctg < 4; ++ctg) {
                v8f acc[2][2] = {};
                const int colBase = col0 + ctg * 32;
                for (int kc = 0; kc < 16; ++kc) {
                    const int d0 = kc * 32;
                    v16bf afr[2];
                    #pragma unroll
                    for (int rt = 0; rt < 2; ++rt) {
                        const int rowA = rt * 16 + laneN;
                        v8bf alo = *(const v8bf*)&resBF[rowA][d0 + koffA];
                        v8bf ahi = *(const v8bf*)&resBF[rowA][d0 + koffA + 8];
                        afr[rt] = __builtin_shufflevector(alo, ahi,
                                  0,1,2,3,4,5,6,7,8,9,10,11,12,13,14,15);
                    }
                    #pragma unroll
                    for (int ct = 0; ct < 2; ++ct) {
                        const int cA = colBase + ct * 16 + laneN;
                        const float* bp = cbq + (size_t)cA * Dd + d0 + koffB;
                        v4f f0 = *(const v4f*)(bp);
                        v4f f1 = *(const v4f*)(bp + 4);
                        v4f f2 = *(const v4f*)(bp + 8);
                        v4f f3 = *(const v4f*)(bp + 12);
                        v16bf bfr;
                        #pragma unroll
                        for (int e = 0; e < 4; ++e) {
                            bfr[e]      = (__bf16)f0[e];
                            bfr[4 + e]  = (__bf16)f1[e];
                            bfr[8 + e]  = (__bf16)f2[e];
                            bfr[12 + e] = (__bf16)f3[e];
                        }
                        #pragma unroll
                        for (int rt = 0; rt < 2; ++rt)
                            acc[rt][ct] = __builtin_amdgcn_wmma_f32_16x16x32_bf16(
                                false, afr[rt], false, bfr, (short)0, acc[rt][ct],
                                false, false);
                    }
                }
                #pragma unroll
                for (int ct = 0; ct < 2; ++ct) {
                    const int cA = colBase + ct * 16 + laneN;
                    const float c2v = c2q[cA];
                    #pragma unroll
                    for (int rt = 0; rt < 2; ++rt)
                        #pragma unroll
                        for (int j = 0; j < 8; ++j) {
                            const float dval = c2v - 2.0f * acc[rt][ct][j];
                            if (dval < minv[rt][j]) { minv[rt][j] = dval; mini[rt][j] = cA; }
                        }
                }
            }
        }

        // ---- intra-wave argmin across the 16 lanes of each half ----
        #pragma unroll
        for (int rt = 0; rt < 2; ++rt)
            #pragma unroll
            for (int j = 0; j < 8; ++j) {
                float v = minv[rt][j]; int bi = mini[rt][j];
                #pragma unroll
                for (int off = 8; off >= 1; off >>= 1) {
                    float ov = __shfl_xor(v,  off, 16);
                    int   oi = __shfl_xor(bi, off, 16);
                    if (ov < v) { v = ov; bi = oi; }
                }
                minv[rt][j] = v; mini[rt][j] = bi;
            }
        if (laneN == 0) {
            #pragma unroll
            for (int rt = 0; rt < 2; ++rt)
                #pragma unroll
                for (int j = 0; j < 8; ++j) {
                    const int row = rt * 16 + halfSel * 8 + j;  // C layout: M=j (+8 hi half)
                    candV[wv][row] = minv[rt][j];
                    candI[wv][row] = mini[rt][j];
                }
        }
        __syncthreads();

        // ---- cross-wave combine; emit index + histogram ----
        if (tid < 32) {
            float bv = candV[0][tid]; int bi = candI[0][tid];
            #pragma unroll
            for (int w2 = 1; w2 < 8; ++w2) {
                const float v = candV[w2][tid];
                const int  i2 = candI[w2][tid];
                if (v < bv) { bv = v; bi = i2; }
            }
            idxLDS[tid] = bi;
            outIdx[(size_t)(n0 + tid) * Qq + q] = (float)bi;
            atomicAdd(&counts[q * Vv + bi], 1.0f);   // exact integer-valued adds
        }
        __syncthreads();

        // ---- fp32 residual update + loss partial (sum of new residual^2) ----
        const int code = idxLDS[r];
        const float* crow = cbq + (size_t)code * Dd;
        float lacc = 0.0f;
        #pragma unroll
        for (int i = 0; i < 64; ++i) {
            const int d = g + (i << 3);
            const float rv = resR[i] - crow[d];
            resR[i] = rv;
            resBF[r][d] = (__bf16)rv;
            lacc += rv * rv;
        }
        #pragma unroll
        for (int off = 16; off >= 1; off >>= 1) lacc += __shfl_xor(lacc, off, 32);
        if (lane == 0) lossW[wv] = lacc;
        __syncthreads();                              // also publishes resBF for next q
        if (tid == 0) {
            float s = 0.0f;
            #pragma unroll
            for (int w2 = 0; w2 < 8; ++w2) s += lossW[w2];
            lossPart[q * NBLK + blockIdx.x] = s;      // deterministic (no atomics)
        }
    }

    // ---- quantized_out = x - final residual (exact) ----
    {
        const float* xb = x    + (size_t)bIdx * Dd * Tt + t0 + r;
        float*       ob = outQ + (size_t)bIdx * Dd * Tt + t0 + r;
        #pragma unroll
        for (int i = 0; i < 64; ++i) {
            const int d = g + (i << 3);
            ob[(size_t)d * Tt] = xb[(size_t)d * Tt] - resR[i];
        }
    }
}

// -------------------------------------------------------------------------
// Finalize: usage / loss / perplexity scalars (single block, deterministic).
// -------------------------------------------------------------------------
__global__ __launch_bounds__(256)
void rvq_finalize(const float* __restrict__ counts,
                  const float* __restrict__ lossPart,
                  float* __restrict__ scal) {
    __shared__ float shA[256];
    __shared__ float shB[256];
    const int tid = threadIdx.x;
    float perpSum = 0.0f, usageSum = 0.0f, lossTot = 0.0f;
    for (int q = 0; q < Qq; ++q) {
        float pl = 0.0f, nz = 0.0f;
        for (int v = tid; v < Vv; v += 256) {
            const float c = counts[q * Vv + v];
            if (c > 0.0f) {
                const float p = c * (1.0f / (float)Nn);
                pl += p * __logf(p);
                nz += 1.0f;
            }
        }
        shA[tid] = pl; shB[tid] = nz;
        __syncthreads();
        for (int s = 128; s > 0; s >>= 1) {
            if (tid < s) { shA[tid] += shA[tid + s]; shB[tid] += shB[tid + s]; }
            __syncthreads();
        }
        if (tid == 0) {
            perpSum  += __expf(-shA[0]);
            usageSum += shB[0] * (100.0f / (float)Vv);
        }
        __syncthreads();
        float ls = 0.0f;
        for (int i = tid; i < NBLK; i += 256) ls += lossPart[q * NBLK + i];
        shA[tid] = ls;
        __syncthreads();
        for (int s = 128; s > 0; s >>= 1) {
            if (tid < s) shA[tid] += shA[tid + s];
            __syncthreads();
        }
        if (tid == 0) lossTot += shA[0];
        __syncthreads();
    }
    if (tid == 0) {
        scal[0] = usageSum / (float)Qq;                                         // usage avg
        scal[1] = (1.0f + BETA) * lossTot / ((float)Qq * (float)Nn * (float)Dd);// loss avg
        scal[2] = perpSum / (float)Qq;                                          // perp avg
    }
}

// -------------------------------------------------------------------------
extern "C" void kernel_launch(void* const* d_in, const int* in_sizes, int n_in,
                              void* d_out, int out_size, void* d_ws, size_t ws_size,
                              hipStream_t stream) {
    const float* x  = (const float*)d_in[0];   // (B, D, T)
    const float* cb = (const float*)d_in[1];   // (Q, V, D)

    const size_t cbElems   = (size_t)Qq * Vv * Dd;       // 3,145,728
    const size_t bfBytes   = cbElems * sizeof(__bf16);   // ~6 MB
    const size_t statFloat = (size_t)Qq * Vv * 2 + (size_t)Qq * NBLK;
    const bool   useBF     = ws_size >= bfBytes + statFloat * sizeof(float);

    __bf16* cbBF = (__bf16*)d_ws;
    float*  ws   = useBF ? (float*)((char*)d_ws + bfBytes) : (float*)d_ws;
    float* c2       = ws;                        // Q*V
    float* counts   = ws + (size_t)Qq * Vv;      // Q*V
    float* lossPart = counts + (size_t)Qq * Vv;  // Q*NBLK

    float* out    = (float*)d_out;
    float* outQ   = out;                               // B*D*T
    float* scal   = out + (size_t)Bb * Dd * Tt;        // usage, loss, perp
    float* outIdx = scal + 3;                          // N*Q indices (as float)

    rvq_zero<<<(Qq * Vv + 255) / 256, 256, 0, stream>>>(counts, Qq * Vv);
    rvq_prep<<<Qq * Vv, 128, 0, stream>>>(cb, c2, cbBF, useBF ? 1 : 0);
    if (useBF) {
        rvq_main<true><<<NBLK, 256, 0, stream>>>(x, cb, cbBF, c2, outQ, outIdx,
                                                 lossPart, counts);
    } else {
        rvq_main<false><<<NBLK, 256, 0, stream>>>(x, cb, cbBF, c2, outQ, outIdx,
                                                  lossPart, counts);
    }
    rvq_finalize<<<1, 256, 0, stream>>>(counts, lossPart, scal);
}
